// SelfAttentionNode2Node_44444321579204
// MI455X (gfx1250) — matrix operand order repaired and checked by reading, not compiled, not run on hardware
//
#include <hip/hip_runtime.h>
#include <hip/hip_bf16.h>

typedef __attribute__((ext_vector_type(16))) __bf16    v16bf;
typedef __attribute__((ext_vector_type(8)))  float     v8f;
typedef __attribute__((ext_vector_type(8)))  unsigned  v8u;

union frag_u { v16bf bf; v8u u; };

// ---------- bf16 helpers ----------
// Pack two f32 -> bf16x2 dword: round-half-up bias then one v_perm_b32.
__device__ __forceinline__ unsigned pkbf(float lo, float hi) {
  union { float f; unsigned u; } a, b; a.f = lo; b.f = hi;
  unsigned al = a.u + 0x8000u;
  unsigned bh = b.u + 0x8000u;
  // dest: byte0,1 = al bytes 2,3 ; byte2,3 = bh bytes 2,3
  return __builtin_amdgcn_perm(bh, al, 0x07060302u);
}
__device__ __forceinline__ __bf16 f2bf(float f) {       // scalar (epilogues only)
  union { float f; unsigned u; } x; x.f = f;
  unsigned r = x.u + 0x7FFFu + ((x.u >> 16) & 1u);
  unsigned short h = (unsigned short)(r >> 16);
  __bf16 o; __builtin_memcpy(&o, &h, 2); return o;
}
__device__ __forceinline__ float bf2f(__bf16 h) {
  unsigned short s; __builtin_memcpy(&s, &h, 2);
  union { unsigned u; float f; } x; x.u = ((unsigned)s) << 16;
  return x.f;
}

// A-fragment (16x32, MxK) from fp32 row-major tile; this file's contiguous k order:
// lane: m = lane&15 ; element e -> K = e + 16*(lane>=16)
__device__ __forceinline__ v16bf fragA_f32(const float* tile, int ld, int lane) {
  const int r  = lane & 15;
  const int kb = (lane >> 4) << 4;         // 0 or 16
  const float* p = tile + (size_t)r * ld + kb;
  frag_u f;
#pragma unroll
  for (int e = 0; e < 4; ++e) f.u[e]     = pkbf(p[2 * e],      p[2 * e + 1]);
#pragma unroll
  for (int e = 0; e < 4; ++e) f.u[e + 4] = pkbf(p[8 + 2 * e], p[8 + 2 * e + 1]);
  return f.bf;
}

// B-fragment (32x16, KxN) where source is N-major (src[n][k]):
// lane: n = lane&15 ; element e -> K = e + 16*(lane>=16)  (contiguous in k)
__device__ __forceinline__ v16bf fragB_nmajor_f32(const float* src, int ld, int lane) {
  const int n  = lane & 15;
  const int kb = (lane >> 4) << 4;         // 0 or 16
  const float* p = src + (size_t)n * ld + kb;
  frag_u f;
#pragma unroll
  for (int e = 0; e < 8; ++e) f.u[e] = pkbf(p[2 * e], p[2 * e + 1]);
  return f.bf;
}

#define WMMA_BF16(A, B, C) \
  __builtin_amdgcn_wmma_f32_16x16x32_bf16(false, (A), false, (B), (short)0, (C), false, false)

// ---------------- constants ----------------
#define BB   2
#define NN   256
#define HID  256
#define HH   8
#define DD   32
#define SCALE 0.17677669529663687f   // 1/sqrt(32)

#define NQKV_FR (8 * 48 * 32 * 16)   // 196608 bf16 elems
#define NWS_FR  (8 * 6 * 32 * 16)    // 24576 bf16 elems

// ---------------- K1: pack weights into B-fragment order (bf16) ----------------
__global__ void pack_kernel(const float* __restrict__ Wq, const float* __restrict__ Wk,
                            const float* __restrict__ Wv, const float* __restrict__ Wsq,
                            const float* __restrict__ Wsk, const float* __restrict__ Wsv,
                            __bf16* __restrict__ frags_qkv, __bf16* __restrict__ frags_ws) {
  int idx = blockIdx.x * blockDim.x + threadIdx.x;
  if (idx < NQKV_FR) {
    int e    = idx & 15;
    int lane = (idx >> 4) & 31;
    int nt   = (idx >> 9) % 48;
    int k0   = idx / (48 * 512);
    int k = k0 * 32 + e + ((lane >> 4) << 4);
    int n = nt * 16 + (lane & 15);           // 0..767
    int mat = n >> 8, c = n & 255;
    const float* W = (mat == 0) ? Wq : (mat == 1) ? Wk : Wv;
    frags_qkv[idx] = f2bf(W[(size_t)k * 256 + c]);
  } else if (idx < NQKV_FR + NWS_FR) {
    int j    = idx - NQKV_FR;
    int e    = j & 15;
    int lane = (j >> 4) & 31;
    int nt   = (j >> 9) % 6;
    int k0   = j / (6 * 512);
    int k = k0 * 32 + e + ((lane >> 4) << 4);
    int n = nt * 16 + (lane & 15);           // 0..95
    int mat = n >> 5, c = n & 31;
    const float* W = (mat == 0) ? Wsq : (mat == 1) ? Wsk : Wsv;
    frags_ws[j] = f2bf(W[(size_t)k * 32 + c]);
  }
}

// ---------------- K2: fused QKV projection (WMMA), head-layout epilogue ----------------
__global__ __launch_bounds__(32)
void qkv_kernel(const float* __restrict__ hidden, const __bf16* __restrict__ frags_qkv,
                const float* __restrict__ bq, const float* __restrict__ bk,
                const float* __restrict__ bv,
                float* __restrict__ q, float* __restrict__ k, float* __restrict__ v,
                float* __restrict__ vT) {
  const int nt   = blockIdx.x;          // 0..47  (768 cols / 16)
  const int m0   = blockIdx.y * 16;     // 0..511 (B*N rows)
  const int lane = threadIdx.x;
  v8f acc = {};
#pragma unroll
  for (int k0 = 0; k0 < 8; ++k0) {
    v16bf fa = fragA_f32(hidden + (size_t)m0 * 256 + k0 * 32, 256, lane);
    v16bf fb = ((const v16bf*)frags_qkv)[(k0 * 48 + nt) * 32 + lane];
    acc = WMMA_BF16(fa, fb, acc);
  }
  const int col = nt * 16 + (lane & 15);       // 0..767
  const int mat = col >> 8;                    // 0=q 1=k 2=v
  const int hid = col & 255;
  const int hh = hid >> 5, dd = hid & 31;
  const float* bvec = (mat == 0) ? bq : (mat == 1) ? bk : bv;
  float* dst        = (mat == 0) ? q  : (mat == 1) ? k  : v;
  const float bias = bvec[hid];
#pragma unroll
  for (int r = 0; r < 8; ++r) {
    int m  = m0 + r + ((lane >> 4) << 3);      // row in [0,512)
    int bb = m >> 8, n = m & 255;
    float val = acc[r] + bias;
    dst[((((size_t)bb * HH + hh) * NN) + n) * DD + dd] = val;
    if (mat == 2) vT[(((size_t)bb * HH + hh) * DD + dd) * NN + n] = val;
  }
}

// ---------------- K3: fused per-(b,i): struct proj (WMMA) + scores + softmax + P@SV ----------------
__global__ __launch_bounds__(256)
void fused_kernel(const float* __restrict__ structure, const float* __restrict__ mask,
                  const float* __restrict__ qbuf, const float* __restrict__ kbuf,
                  const float* __restrict__ bsq, const float* __restrict__ bsk,
                  const float* __restrict__ bsv, const __bf16* __restrict__ frags_ws,
                  float* __restrict__ probs, float* __restrict__ ctxB) {
  extern __shared__ char smem[];
  __bf16* sbuf   = (__bf16*)smem;                       // [256][96] SQ|SK|SV  (48 KB)
  float*  S      = (float*)(smem + 256 * 96 * 2);       // [8][256] probs       (8 KB)
  float*  kb_lds = S + 8 * 256;                         // [8][32]  k_i         (1 KB)

  const int bi = blockIdx.x;                            // 0..511
  const int b = bi >> 8, i = bi & 255;
  const int t = threadIdx.x, wave = t >> 5, lane = t & 31;

  // preload k[b,:,i,:] into LDS (one float per thread)
  kb_lds[t] = kbuf[((((size_t)b * HH + (t >> 5)) * NN) + i) * DD + (t & 31)];

  // ---- Phase 1: SQ|SK|SV = structure[b,i] (256x256) @ Ws (256x96), WMMA -> LDS bf16
  const float* Sslice = structure + (size_t)bi * NN * HID;
  for (int tile = wave; tile < 96; tile += 8) {
    const int mt = tile / 6, nt = tile % 6;
    if (tile + 8 < 96)
      __builtin_prefetch(Sslice + (size_t)((tile + 8) / 6) * 16 * HID + (lane << 5), 0, 0);
    v8f acc = {};
#pragma unroll
    for (int k0 = 0; k0 < 8; ++k0) {
      v16bf fa = fragA_f32(Sslice + (size_t)(mt * 16) * HID + k0 * 32, HID, lane);
      v16bf fb = ((const v16bf*)frags_ws)[(k0 * 6 + nt) * 32 + lane];
      acc = WMMA_BF16(fa, fb, acc);
    }
    const int col = nt * 16 + (lane & 15);              // 0..95
    const float bias = (col < 32) ? bsq[col] : (col < 64) ? bsk[col - 32] : bsv[col - 64];
#pragma unroll
    for (int r = 0; r < 8; ++r) {
      int j = mt * 16 + r + ((lane >> 4) << 3);
      sbuf[(size_t)j * 96 + col] = f2bf(acc[r] + bias);
    }
  }
  __syncthreads();

  // ---- Phase 2: scores[h,j] = (k_i[h]+sk_j).(q[h,j]+sq_j); softmax over j (wave == h)
  {
    const int h = wave;
    const float* qbase = qbuf + (((size_t)b * HH + h) * NN) * DD;
    float vals[8];
    float mx = -3.0e38f;
#pragma unroll
    for (int jj = 0; jj < 8; ++jj) {
      const int j = lane + jj * 32;
      const float*  qrow = qbase + (size_t)j * DD;
      const __bf16* srow = sbuf + (size_t)j * 96;
      float a = 0.f;
#pragma unroll
      for (int d0 = 0; d0 < 32; ++d0)
        a += (kb_lds[h * 32 + d0] + bf2f(srow[32 + d0])) * (qrow[d0] + bf2f(srow[d0]));
      a = a * SCALE + mask[(size_t)b * NN * NN + (size_t)i * NN + j];
      vals[jj] = a;
      mx = fmaxf(mx, a);
    }
#pragma unroll
    for (int off = 16; off > 0; off >>= 1) mx = fmaxf(mx, __shfl_xor(mx, off, 32));
    float sum = 0.f;
#pragma unroll
    for (int jj = 0; jj < 8; ++jj) { vals[jj] = __expf(vals[jj] - mx); sum += vals[jj]; }
#pragma unroll
    for (int off = 16; off > 0; off >>= 1) sum += __shfl_xor(sum, off, 32);
    const float inv = 1.0f / sum;
#pragma unroll
    for (int jj = 0; jj < 8; ++jj) {
      const int j = lane + jj * 32;
      const float p = vals[jj] * inv;
      S[h * 256 + j] = p;
      probs[((((size_t)b * HH + h) * NN) + i) * NN + j] = p;
    }
  }
  __syncthreads();

  // ---- Phase 3: ctxB[h,:] = P (8x256, padded to 16) @ SV (256x32), WMMA (waves 0..1)
  if (wave < 2) {
    const int nt = wave;                                // d-tile: 0 or 1
    v8f acc = {};
#pragma unroll
    for (int k0 = 0; k0 < 8; ++k0) {
      frag_u fa;
      const int m  = lane & 15;
      const int ka = k0 * 32 + ((lane >> 4) << 4);
      if (m < 8) {
        const float* p = S + m * 256 + ka;
#pragma unroll
        for (int e = 0; e < 4; ++e) {
          fa.u[e]     = pkbf(p[2 * e],      p[2 * e + 1]);
          fa.u[e + 4] = pkbf(p[8 + 2 * e], p[8 + 2 * e + 1]);
        }
      } else {
#pragma unroll
        for (int e = 0; e < 8; ++e) fa.u[e] = 0u;
      }
      v16bf fb;
      const int n  = nt * 16 + (lane & 15);
      const int jb = k0 * 32 + ((lane >> 4) << 4);
#pragma unroll
      for (int e = 0; e < 16; ++e) fb[e] = sbuf[(size_t)(jb + e) * 96 + 64 + n];
      acc = WMMA_BF16(fa.bf, fb, acc);
    }
    const int d = nt * 16 + (lane & 15);
#pragma unroll
    for (int r = 0; r < 8; ++r) {
      const int h = r + ((lane >> 4) << 3);
      if (h < 8)
        ctxB[((((size_t)b * HH + h) * NN) + i) * DD + d] = acc[r];
    }
  }
}

// ---------------- K4: ctxA = P @ V per (b,h), add ctxB, transposed store ----------------
__global__ __launch_bounds__(64)
void ctx_kernel(const float* __restrict__ probs, const float* __restrict__ vT,
                const float* __restrict__ ctxB, float* __restrict__ out) {
  const int nt   = threadIdx.x >> 5;        // d-tile 0/1
  const int lane = threadIdx.x & 31;
  const int bh = blockIdx.y;                // 0..15
  const int i0 = blockIdx.x * 16;           // 0..255
  const float* prow   = probs + (size_t)bh * NN * NN + (size_t)i0 * NN;   // ld 256
  const float* vtbase = vT + (size_t)bh * DD * NN;                        // (d, j) ld 256
  v8f acc = {};
#pragma unroll
  for (int k0 = 0; k0 < 8; ++k0) {
    v16bf fa = fragA_f32(prow + k0 * 32, NN, lane);
    v16bf fb = fragB_nmajor_f32(vtbase + (size_t)(nt * 16) * NN + k0 * 32, NN, lane);
    acc = WMMA_BF16(fa, fb, acc);
  }
  const int b = bh >> 3, h = bh & 7;
  const int d = nt * 16 + (lane & 15);
#pragma unroll
  for (int r = 0; r < 8; ++r) {
    const int i = i0 + r + ((lane >> 4) << 3);
    const float val = acc[r] + ctxB[((size_t)bh * NN + i) * DD + d];
    out[((size_t)b * NN + i) * HID + h * DD + d] = val;
  }
}

// ---------------- host launcher ----------------
extern "C" void kernel_launch(void* const* d_in, const int* in_sizes, int n_in,
                              void* d_out, int out_size, void* d_ws, size_t ws_size,
                              hipStream_t stream) {
  (void)in_sizes; (void)n_in; (void)out_size; (void)ws_size;
  const float* hidden    = (const float*)d_in[0];
  const float* mask      = (const float*)d_in[1];
  const float* structure = (const float*)d_in[2];
  const float* Wq  = (const float*)d_in[3];
  const float* bq  = (const float*)d_in[4];
  const float* Wk  = (const float*)d_in[5];
  const float* bk  = (const float*)d_in[6];
  const float* Wv  = (const float*)d_in[7];
  const float* bv  = (const float*)d_in[8];
  const float* Wsq = (const float*)d_in[9];
  const float* bsq = (const float*)d_in[10];
  const float* Wsk = (const float*)d_in[11];
  const float* bsk = (const float*)d_in[12];
  const float* Wsv = (const float*)d_in[13];
  const float* bsv = (const float*)d_in[14];
  float* out = (float*)d_out;

  char* ws = (char*)d_ws;
  float* q     = (float*)ws; ws += (size_t)131072 * 4;
  float* k     = (float*)ws; ws += (size_t)131072 * 4;
  float* v     = (float*)ws; ws += (size_t)131072 * 4;
  float* vT    = (float*)ws; ws += (size_t)131072 * 4;
  float* ctxB  = (float*)ws; ws += (size_t)131072 * 4;
  float* probs = (float*)ws; ws += (size_t)1048576 * 4;
  __bf16* frags_qkv = (__bf16*)ws; ws += (size_t)NQKV_FR * 2;
  __bf16* frags_ws  = (__bf16*)ws; ws += (size_t)NWS_FR * 2;

  const int total_pack = NQKV_FR + NWS_FR;     // 221184
  pack_kernel<<<dim3((total_pack + 255) / 256), dim3(256), 0, stream>>>(
      Wq, Wk, Wv, Wsq, Wsk, Wsv, frags_qkv, frags_ws);

  qkv_kernel<<<dim3(48, 32), dim3(32), 0, stream>>>(
      hidden, frags_qkv, bq, bk, bv, q, k, v, vT);

  const size_t smem = (size_t)256 * 96 * 2 + 8 * 256 * 4 + 256 * 4;   // 58368 B
  fused_kernel<<<dim3(512), dim3(256), smem, stream>>>(
      structure, mask, q, k, bsq, bsk, bsv, frags_ws, probs, ctxB);

  ctx_kernel<<<dim3(16, 16), dim3(64), 0, stream>>>(probs, vT, ctxB, out);
}
